// BertLayer_26714696581657
// MI455X (gfx1250) — matrix operand order repaired and checked by reading, not compiled, not run on hardware
//
#include <hip/hip_runtime.h>
#include <math.h>

typedef __attribute__((ext_vector_type(16))) __bf16 bf16x16;
typedef __attribute__((ext_vector_type(8)))  float  f32x8;

#define DEV __device__ __forceinline__

constexpr int Bc = 8, Sc = 512, Hc = 768, NHc = 12, DHc = 64, Ic = 3072, Ec = 8;
constexpr int NTOK = Bc * Sc;      // 4096
constexpr int LDP  = 40;           // LDS pitch (ushorts) for 32-wide K tiles (multiple of 8)
constexpr float EPSc = 1e-12f;

DEV unsigned short f2bf(float f) {
  unsigned u = __builtin_bit_cast(unsigned, f);
  u += 0x7FFFu + ((u >> 16) & 1u);
  return (unsigned short)(u >> 16);
}
DEV float bf2f(unsigned short h) {
  return __builtin_bit_cast(float, (unsigned)h << 16);
}

union FragU { bf16x16 v; uint4 q[2]; };

// A-operand fragment (16xK=32 bf16), ISA layout: lanes 0-15 M=0..15 K0..7 & K16..23,
// lanes 16-31 same rows, K8..15 & K24..31. Tile is row-major [16][ld].
DEV bf16x16 load_frag_a(const unsigned short* p, int ld) {
  int lane = threadIdx.x & 31;
  const unsigned short* b = p + (lane & 15) * ld + ((lane >> 4) << 3);
  FragU u;
  u.q[0] = *(const uint4*)(b);
  u.q[1] = *(const uint4*)(b + 16);
  return u.v;
}
// B-operand fragment (K=32 x N=16 bf16) from N-major tile Bt[n][k]:
// lanes 0-15 hold K0..7 & K16..23, lanes 16-31 hold K8..15 & K24..31 for column n = lane&15.
DEV bf16x16 load_frag_b(const unsigned short* p, int ld) {
  int lane = threadIdx.x & 31;
  const unsigned short* b = p + (lane & 15) * ld + ((lane >> 4) << 3);
  FragU u;
  u.q[0] = *(const uint4*)(b);
  u.q[1] = *(const uint4*)(b + 16);
  return u.v;
}

DEV f32x8 wmma_bf16(bf16x16 a, bf16x16 b, f32x8 c) {
  return __builtin_amdgcn_wmma_f32_16x16x32_bf16(false, a, false, b, (short)0, c,
                                                 false, false);
}
DEV f32x8 zero8() {
  f32x8 z;
#pragma unroll
  for (int i = 0; i < 8; ++i) z[i] = 0.f;
  return z;
}

// -------- CDNA5 async global->LDS copy (ASYNCcnt-tracked, bypasses VGPRs) --------
// Flat shared-pointer low 32 bits == LDS byte offset (LDS aperture: addr[31:0]).
DEV unsigned lds_off(const void* p) { return (unsigned)(unsigned long long)p; }

DEV void async_b128(unsigned ldsdst, const void* g) {
  asm volatile("global_load_async_to_lds_b128 %0, %1, off"
               :: "v"(ldsdst), "v"((unsigned long long)g)
               : "memory");
}
DEV void wait_async0() {
  asm volatile("s_wait_asynccnt 0x0" ::: "memory");
}

// ---------------- cooperative async LDS tile loads (256 threads) ----------------
DEV void a_tile_rows128(unsigned short* dst, const unsigned short* src, int ldsrc,
                        long row0, int k0) {
  int t = threadIdx.x;
  int row = t >> 1;
  int h = (t & 1) << 4;
  const unsigned short* g = src + (long)(row0 + row) * ldsrc + k0 + h;
  unsigned d = lds_off(dst + row * LDP + h);
  async_b128(d, g);
  async_b128(d + 16, g + 8);
}
DEV void a_tile_rows64(unsigned short* dst, const unsigned short* src, int ldsrc,
                       long row0, int k0) {
  int t = threadIdx.x;
  int row = t >> 2;
  int h = (t & 3) << 3;
  async_b128(lds_off(dst + row * LDP + h), src + (long)(row0 + row) * ldsrc + k0 + h);
}
DEV void a_tile_gather128(unsigned short* dst, const unsigned short* src, int ldsrc,
                          const int* rows, int cnt, int base, int k0) {
  int t = threadIdx.x;
  int row = t >> 1;
  int h = (t & 1) << 4;
  int gi = base + row;
  int tok = (gi < cnt) ? rows[gi] : 0;
  const unsigned short* g = src + (long)tok * ldsrc + k0 + h;
  unsigned d = lds_off(dst + row * LDP + h);
  async_b128(d, g);
  async_b128(d + 16, g + 8);
}
DEV void a_tile_clamp128(unsigned short* dst, const unsigned short* src, int ldsrc,
                         int base, int cnt, int k0) {
  int t = threadIdx.x;
  int row = t >> 1;
  int h = (t & 1) << 4;
  int r = base + row;
  if (r >= cnt) r = cnt - 1;
  const unsigned short* g = src + (long)r * ldsrc + k0 + h;
  unsigned d = lds_off(dst + row * LDP + h);
  async_b128(d, g);
  async_b128(d + 16, g + 8);
}

// ---------------- per-K-step WMMA blocks ----------------
DEV void mm_step128(const unsigned short* sA, const unsigned short* sB, int wm, int wn,
                    f32x8 c[2][4]) {
  bf16x16 a0 = load_frag_a(sA + (wm * 32) * LDP, LDP);
  bf16x16 a1 = load_frag_a(sA + (wm * 32 + 16) * LDP, LDP);
#pragma unroll
  for (int j = 0; j < 4; ++j) {
    bf16x16 b = load_frag_b(sB + (wn * 64 + j * 16) * LDP, LDP);
    c[0][j] = wmma_bf16(a0, b, c[0][j]);
    c[1][j] = wmma_bf16(a1, b, c[1][j]);
  }
}
DEV void mm_step64x2(const unsigned short* sA, const unsigned short* sBu,
                     const unsigned short* sBn, int wm, int wn, f32x8 cu[2][2],
                     f32x8 cn[2][2]) {
  bf16x16 a0 = load_frag_a(sA + (wm * 32) * LDP, LDP);
  bf16x16 a1 = load_frag_a(sA + (wm * 32 + 16) * LDP, LDP);
#pragma unroll
  for (int j = 0; j < 2; ++j) {
    bf16x16 bu = load_frag_b(sBu + (wn * 32 + j * 16) * LDP, LDP);
    bf16x16 bn = load_frag_b(sBn + (wn * 32 + j * 16) * LDP, LDP);
    cu[0][j] = wmma_bf16(a0, bu, cu[0][j]);
    cu[1][j] = wmma_bf16(a1, bu, cu[1][j]);
    cn[0][j] = wmma_bf16(a0, bn, cn[0][j]);
    cn[1][j] = wmma_bf16(a1, bn, cn[1][j]);
  }
}
DEV void mm_step64(const unsigned short* sA, const unsigned short* sB, int wm, int wn,
                   f32x8 c[2][2]) {
  bf16x16 a0 = load_frag_a(sA + (wm * 32) * LDP, LDP);
  bf16x16 a1 = load_frag_a(sA + (wm * 32 + 16) * LDP, LDP);
#pragma unroll
  for (int j = 0; j < 2; ++j) {
    bf16x16 b = load_frag_b(sB + (wn * 32 + j * 16) * LDP, LDP);
    c[0][j] = wmma_bf16(a0, b, c[0][j]);
    c[1][j] = wmma_bf16(a1, b, c[1][j]);
  }
}

// ---------------- misc small kernels ----------------
__global__ void __launch_bounds__(256) k_zero(float* moe, int* counts) {
  long i = (long)blockIdx.x * 256 + threadIdx.x;
  if (i < (long)NTOK * Hc) moe[i] = 0.f;
  if (i < Ec) counts[i] = 0;
}
__global__ void __launch_bounds__(256) k_cvt(const float* __restrict__ src,
                                             unsigned short* __restrict__ dst, long n) {
  long i = (long)blockIdx.x * 256 + threadIdx.x;
  if (i < n) dst[i] = f2bf(src[i]);
}
// transpose-convert: src f32 [K][N] -> dst bf16 [N][K]; grid (N/32, K/32, batch)
__global__ void __launch_bounds__(256) k_tcvt(const float* __restrict__ src,
                                              unsigned short* __restrict__ dst, int K,
                                              int N, long sstride, long dstride) {
  __shared__ __align__(16) unsigned short tile[32][33];
  const float* s = src + (long)blockIdx.z * sstride;
  unsigned short* d = dst + (long)blockIdx.z * dstride;
  int n0 = blockIdx.x * 32, k0 = blockIdx.y * 32;
  int tx = threadIdx.x & 31, ty = threadIdx.x >> 5;
#pragma unroll
  for (int i = 0; i < 32; i += 8)
    tile[ty + i][tx] = f2bf(s[(long)(k0 + ty + i) * N + n0 + tx]);
  __syncthreads();
#pragma unroll
  for (int i = 0; i < 32; i += 8)
    d[(long)(n0 + ty + i) * K + k0 + tx] = tile[tx][ty + i];
}

// ---------------- QKV GEMM: x16[4096x768] @ Wt[768x768] + bias -> [B,NH,S,DH] bf16 ----
__global__ void __launch_bounds__(256) k_gemm_qkv(const unsigned short* __restrict__ A,
                                                  const unsigned short* __restrict__ Bt,
                                                  const float* __restrict__ bias,
                                                  unsigned short* __restrict__ out) {
  __shared__ __align__(16) unsigned short sA[2][128 * LDP];
  __shared__ __align__(16) unsigned short sB[2][128 * LDP];
  int n0 = blockIdx.x * 128;
  long m0 = (long)blockIdx.y * 128;
  int wave = threadIdx.x >> 5, lane = threadIdx.x & 31;
  int wm = wave & 3, wn = wave >> 2;
  f32x8 c[2][4];
#pragma unroll
  for (int i = 0; i < 2; ++i)
#pragma unroll
    for (int j = 0; j < 4; ++j) c[i][j] = zero8();
  a_tile_rows128(sA[0], A, Hc, m0, 0);
  a_tile_rows128(sB[0], Bt, Hc, n0, 0);
  wait_async0();
  __syncthreads();
  const int nk = Hc / 32;   // 24, even
  int kt = 0;
  for (; kt + 2 < nk; kt += 2) {
    a_tile_rows128(sA[1], A, Hc, m0, (kt + 1) * 32);
    a_tile_rows128(sB[1], Bt, Hc, n0, (kt + 1) * 32);
    mm_step128(sA[0], sB[0], wm, wn, c);
    wait_async0();
    __syncthreads();
    a_tile_rows128(sA[0], A, Hc, m0, (kt + 2) * 32);
    a_tile_rows128(sB[0], Bt, Hc, n0, (kt + 2) * 32);
    mm_step128(sA[1], sB[1], wm, wn, c);
    wait_async0();
    __syncthreads();
  }
  a_tile_rows128(sA[1], A, Hc, m0, (kt + 1) * 32);
  a_tile_rows128(sB[1], Bt, Hc, n0, (kt + 1) * 32);
  mm_step128(sA[0], sB[0], wm, wn, c);
  wait_async0();
  __syncthreads();
  mm_step128(sA[1], sB[1], wm, wn, c);
  int hl = lane >> 4, ll = lane & 15;
#pragma unroll
  for (int i = 0; i < 2; ++i)
#pragma unroll
    for (int j = 0; j < 4; ++j)
#pragma unroll
      for (int r = 0; r < 8; ++r) {
        int m = (int)m0 + wm * 32 + i * 16 + r + 8 * hl;
        int n = n0 + wn * 64 + j * 16 + ll;
        float v = c[i][j][r] + bias[n];
        int head = n >> 6, dd = n & 63;
        int bz = m >> 9, ss = m & 511;
        out[((((long)bz * NHc + head) * Sc + ss) << 6) + dd] = f2bf(v);
      }
}

// ---------------- Wo GEMM: ctx[4096x768] @ WoT -> f32 tmp ----------------
__global__ void __launch_bounds__(256) k_gemm_wo(const unsigned short* __restrict__ A,
                                                 const unsigned short* __restrict__ Bt,
                                                 float* __restrict__ out) {
  __shared__ __align__(16) unsigned short sA[2][128 * LDP];
  __shared__ __align__(16) unsigned short sB[2][128 * LDP];
  int n0 = blockIdx.x * 128;
  long m0 = (long)blockIdx.y * 128;
  int wave = threadIdx.x >> 5, lane = threadIdx.x & 31;
  int wm = wave & 3, wn = wave >> 2;
  f32x8 c[2][4];
#pragma unroll
  for (int i = 0; i < 2; ++i)
#pragma unroll
    for (int j = 0; j < 4; ++j) c[i][j] = zero8();
  a_tile_rows128(sA[0], A, Hc, m0, 0);
  a_tile_rows128(sB[0], Bt, Hc, n0, 0);
  wait_async0();
  __syncthreads();
  const int nk = Hc / 32;
  int kt = 0;
  for (; kt + 2 < nk; kt += 2) {
    a_tile_rows128(sA[1], A, Hc, m0, (kt + 1) * 32);
    a_tile_rows128(sB[1], Bt, Hc, n0, (kt + 1) * 32);
    mm_step128(sA[0], sB[0], wm, wn, c);
    wait_async0();
    __syncthreads();
    a_tile_rows128(sA[0], A, Hc, m0, (kt + 2) * 32);
    a_tile_rows128(sB[0], Bt, Hc, n0, (kt + 2) * 32);
    mm_step128(sA[1], sB[1], wm, wn, c);
    wait_async0();
    __syncthreads();
  }
  a_tile_rows128(sA[1], A, Hc, m0, (kt + 1) * 32);
  a_tile_rows128(sB[1], Bt, Hc, n0, (kt + 1) * 32);
  mm_step128(sA[0], sB[0], wm, wn, c);
  wait_async0();
  __syncthreads();
  mm_step128(sA[1], sB[1], wm, wn, c);
  int hl = lane >> 4, ll = lane & 15;
#pragma unroll
  for (int i = 0; i < 2; ++i)
#pragma unroll
    for (int j = 0; j < 4; ++j)
#pragma unroll
      for (int r = 0; r < 8; ++r) {
        long m = m0 + wm * 32 + i * 16 + r + 8 * hl;
        int n = n0 + wn * 64 + j * 16 + ll;
        out[m * Hc + n] = c[i][j][r];
      }
}

// ---------------- flash attention: grid (S/128, NH, B), 256 thr ----------------
__global__ void __launch_bounds__(256) k_attn(const unsigned short* __restrict__ Q,
                                              const unsigned short* __restrict__ K,
                                              const unsigned short* __restrict__ V,
                                              unsigned short* __restrict__ ctx) {
  __shared__ __align__(16) unsigned short sQ[128 * 72];
  __shared__ __align__(16) unsigned short sK[64 * 72];
  __shared__ __align__(16) unsigned short sVt[64 * 72];
  __shared__ __align__(16) unsigned short sP[8 * 16 * 72];
  int q0 = blockIdx.x * 128;
  int hh = blockIdx.y;
  int bz = blockIdx.z;
  const unsigned short* Qh = Q + ((long)(bz * NHc + hh) * Sc) * DHc;
  const unsigned short* Kh = K + ((long)(bz * NHc + hh) * Sc) * DHc;
  const unsigned short* Vh = V + ((long)(bz * NHc + hh) * Sc) * DHc;
  int t = threadIdx.x;
  int wave = t >> 5, lane = t & 31;
  int hl = lane >> 4, ll = lane & 15;
  for (int i = t; i < 1024; i += 256) {
    int row = i >> 3, c8 = (i & 7) << 3;
    async_b128(lds_off(&sQ[row * 72 + c8]), &Qh[(long)(q0 + row) * DHc + c8]);
  }
  f32x8 acc[4];
#pragma unroll
  for (int j = 0; j < 4; ++j) acc[j] = zero8();
  float mrow[8], lrow[8];
#pragma unroll
  for (int r = 0; r < 8; ++r) { mrow[r] = -1e30f; lrow[r] = 0.f; }
  unsigned short* pb = sP + wave * (16 * 72);
  for (int kv0 = 0; kv0 < Sc; kv0 += 64) {
    __syncthreads();
    for (int i = t; i < 512; i += 256) {
      int row = i >> 3, c8 = (i & 7) << 3;
      async_b128(lds_off(&sK[row * 72 + c8]), &Kh[(long)(kv0 + row) * DHc + c8]);
    }
    for (int i = t; i < 4096; i += 256) {
      int kvr = i >> 6, dd = i & 63;
      sVt[dd * 72 + kvr] = Vh[(long)(kv0 + kvr) * DHc + dd];
    }
    wait_async0();
    __syncthreads();
    bf16x16 a0 = load_frag_a(sQ + (wave * 16) * 72, 72);
    bf16x16 a1 = load_frag_a(sQ + (wave * 16) * 72 + 32, 72);
    f32x8 s[4];
#pragma unroll
    for (int j = 0; j < 4; ++j) {
      bf16x16 b0 = load_frag_b(sK + (j * 16) * 72, 72);
      bf16x16 b1 = load_frag_b(sK + (j * 16) * 72 + 32, 72);
      f32x8 cc = zero8();
      cc = wmma_bf16(a0, b0, cc);
      cc = wmma_bf16(a1, b1, cc);
#pragma unroll
      for (int r = 0; r < 8; ++r) cc[r] *= 0.125f;
      s[j] = cc;
    }
#pragma unroll
    for (int r = 0; r < 8; ++r) {
      float mx = fmaxf(fmaxf(s[0][r], s[1][r]), fmaxf(s[2][r], s[3][r]));
      for (int off = 1; off < 16; off <<= 1) mx = fmaxf(mx, __shfl_xor(mx, off, 32));
      float mn = fmaxf(mrow[r], mx);
      float alpha = __expf(mrow[r] - mn);
      mrow[r] = mn;
      float rs = 0.f;
#pragma unroll
      for (int j = 0; j < 4; ++j) {
        float p = __expf(s[j][r] - mn);
        s[j][r] = p;
        rs += p;
      }
      for (int off = 1; off < 16; off <<= 1) rs += __shfl_xor(rs, off, 32);
      lrow[r] = lrow[r] * alpha + rs;
#pragma unroll
      for (int j = 0; j < 4; ++j) acc[j][r] *= alpha;
    }
#pragma unroll
    for (int j = 0; j < 4; ++j)
#pragma unroll
      for (int r = 0; r < 8; ++r)
        pb[(r + 8 * hl) * 72 + j * 16 + ll] = f2bf(s[j][r]);
    bf16x16 pa0 = load_frag_a(pb, 72);
    bf16x16 pa1 = load_frag_a(pb + 32, 72);
#pragma unroll
    for (int j = 0; j < 4; ++j) {
      bf16x16 bv0 = load_frag_b(sVt + (j * 16) * 72, 72);
      bf16x16 bv1 = load_frag_b(sVt + (j * 16) * 72 + 32, 72);
      acc[j] = wmma_bf16(pa0, bv0, acc[j]);
      acc[j] = wmma_bf16(pa1, bv1, acc[j]);
    }
  }
#pragma unroll
  for (int j = 0; j < 4; ++j)
#pragma unroll
    for (int r = 0; r < 8; ++r) {
      int row = q0 + wave * 16 + r + 8 * hl;
      int dd = j * 16 + ll;
      float v = acc[j][r] / lrow[r];
      ctx[((long)(bz * Sc + row)) * Hc + hh * DHc + dd] = f2bf(v);
    }
}

// ---------------- block reduce helper ----------------
DEV float block_sum(float v, float* red) {
  red[threadIdx.x] = v;
  __syncthreads();
  for (int s = 128; s > 0; s >>= 1) {
    if ((int)threadIdx.x < s) red[threadIdx.x] += red[threadIdx.x + s];
    __syncthreads();
  }
  float r = red[0];
  __syncthreads();
  return r;
}

// ---------------- fused LN(attn residual) + LN(ffn) ----------------
__global__ void __launch_bounds__(256) k_ln2(const float* __restrict__ tmp,
                                             const float* __restrict__ x,
                                             const float* __restrict__ bo,
                                             const float* __restrict__ g1,
                                             const float* __restrict__ bb1,
                                             const float* __restrict__ g2,
                                             const float* __restrict__ bb2,
                                             float* __restrict__ ao,
                                             unsigned short* __restrict__ xl) {
  __shared__ float red[256];
  long tk = blockIdx.x;
  float v[3];
  float s = 0.f;
#pragma unroll
  for (int i = 0; i < 3; ++i) {
    int c = threadIdx.x + i * 256;
    v[i] = tmp[tk * Hc + c] + bo[c] + x[tk * Hc + c];
    s += v[i];
  }
  s = block_sum(s, red);
  float mean = s * (1.f / 768.f);
  float ss = 0.f;
#pragma unroll
  for (int i = 0; i < 3; ++i) { float d = v[i] - mean; ss += d * d; }
  ss = block_sum(ss, red);
  float inv = rsqrtf(ss * (1.f / 768.f) + EPSc);
  float y[3];
  float s2 = 0.f;
#pragma unroll
  for (int i = 0; i < 3; ++i) {
    int c = threadIdx.x + i * 256;
    y[i] = (v[i] - mean) * inv * g1[c] + bb1[c];
    ao[tk * Hc + c] = y[i];
    s2 += y[i];
  }
  s2 = block_sum(s2, red);
  float mean2 = s2 * (1.f / 768.f);
  float ss2 = 0.f;
#pragma unroll
  for (int i = 0; i < 3; ++i) { float d = y[i] - mean2; ss2 += d * d; }
  ss2 = block_sum(ss2, red);
  float inv2 = rsqrtf(ss2 * (1.f / 768.f) + EPSc);
#pragma unroll
  for (int i = 0; i < 3; ++i) {
    int c = threadIdx.x + i * 256;
    xl[tk * Hc + c] = f2bf((y[i] - mean2) * inv2 * g2[c] + bb2[c]);
  }
}

// ---------------- router + top2 + counts; grid 512 x 256 (1 wave per token) ----
__global__ void __launch_bounds__(256) k_router(const unsigned short* __restrict__ xl,
                                                const float* __restrict__ Wr,
                                                const float* __restrict__ br,
                                                float* __restrict__ rlog,
                                                int* __restrict__ top2i,
                                                float* __restrict__ top2w,
                                                int* __restrict__ counts) {
  int wave = threadIdx.x >> 5, lane = threadIdx.x & 31;
  int tk = blockIdx.x * 8 + wave;
  float acc[8];
#pragma unroll
  for (int e = 0; e < 8; ++e) acc[e] = 0.f;
  for (int h = lane; h < Hc; h += 32) {
    float xv = bf2f(xl[(long)tk * Hc + h]);
    const float* wr = Wr + h * Ec;
#pragma unroll
    for (int e = 0; e < 8; ++e) acc[e] += xv * wr[e];
  }
  for (int off = 16; off; off >>= 1)
#pragma unroll
    for (int e = 0; e < 8; ++e) acc[e] += __shfl_xor(acc[e], off, 32);
  if (lane == 0) {
    float lg[8], p[8];
    float mx = -1e30f;
#pragma unroll
    for (int e = 0; e < 8; ++e) {
      lg[e] = acc[e] + br[e];
      rlog[(long)tk * Ec + e] = lg[e];
      mx = fmaxf(mx, lg[e]);
    }
#pragma unroll
    for (int e = 0; e < 8; ++e) p[e] = __expf(lg[e] - mx);
    int e0 = 0;
#pragma unroll
    for (int e = 1; e < 8; ++e) if (p[e] > p[e0]) e0 = e;
    int e1 = (e0 == 0) ? 1 : 0;
#pragma unroll
    for (int e = 0; e < 8; ++e) if (e != e0 && p[e] > p[e1]) e1 = e;
    float denom = p[e0] + p[e1];
    top2i[tk * 2] = e0;
    top2i[tk * 2 + 1] = e1;
    top2w[tk * 2] = p[e0] / denom;
    top2w[tk * 2 + 1] = p[e1] / denom;
    atomicAdd(&counts[e0], 1);
    atomicAdd(&counts[e1], 1);
  }
}

__global__ void k_scan(const int* __restrict__ counts, int* __restrict__ segoff,
                       int* __restrict__ cursor) {
  if (threadIdx.x == 0) {
    int o = 0;
    for (int e = 0; e < Ec; ++e) { segoff[e] = o; o += counts[e]; }
    segoff[Ec] = o;
  }
  if ((int)threadIdx.x < Ec) cursor[threadIdx.x] = 0;
}

__global__ void __launch_bounds__(256) k_scatter(const int* __restrict__ top2i,
                                                 const float* __restrict__ top2w,
                                                 const int* __restrict__ segoff,
                                                 int* __restrict__ cursor,
                                                 int* __restrict__ segtok,
                                                 float* __restrict__ segw) {
  int tk = blockIdx.x * 256 + threadIdx.x;
  if (tk >= NTOK) return;
#pragma unroll
  for (int k = 0; k < 2; ++k) {
    int e = top2i[tk * 2 + k];
    int slot = atomicAdd(&cursor[e], 1);
    segtok[segoff[e] + slot] = tk;
    segw[segoff[e] + slot] = top2w[tk * 2 + k];
  }
}

// ---------------- expert up/new fused GEMM; grid (I/64, 32, E) ----------------
__global__ void __launch_bounds__(256) k_expert_upnew(
    const unsigned short* __restrict__ XL, const unsigned short* __restrict__ WupT,
    const unsigned short* __restrict__ WnewT, const float* __restrict__ b_up,
    const float* __restrict__ b_new, const int* __restrict__ counts,
    const int* __restrict__ segoff, const int* __restrict__ segtok,
    unsigned short* __restrict__ hbuf) {
  int e = blockIdx.z;
  int cnt = counts[e];
  int mbase = blockIdx.y * 128;
  if (mbase >= cnt) return;
  __shared__ __align__(16) unsigned short sA[2][128 * LDP];
  __shared__ __align__(16) unsigned short sBu[2][64 * LDP];
  __shared__ __align__(16) unsigned short sBn[2][64 * LDP];
  int off = segoff[e];
  const int* rows = segtok + off;
  int n0 = blockIdx.x * 64;
  const unsigned short* Bu = WupT + (long)e * Ic * Hc;
  const unsigned short* Bn = WnewT + (long)e * Ic * Hc;
  int wave = threadIdx.x >> 5, lane = threadIdx.x & 31;
  int wm = wave & 3, wn = wave >> 2;
  f32x8 cu[2][2], cn[2][2];
#pragma unroll
  for (int i = 0; i < 2; ++i)
#pragma unroll
    for (int j = 0; j < 2; ++j) { cu[i][j] = zero8(); cn[i][j] = zero8(); }
  a_tile_gather128(sA[0], XL, Hc, rows, cnt, mbase, 0);
  a_tile_rows64(sBu[0], Bu, Hc, n0, 0);
  a_tile_rows64(sBn[0], Bn, Hc, n0, 0);
  wait_async0();
  __syncthreads();
  const int nk = Hc / 32;   // 24, even
  int kt = 0;
  for (; kt + 2 < nk; kt += 2) {
    a_tile_gather128(sA[1], XL, Hc, rows, cnt, mbase, (kt + 1) * 32);
    a_tile_rows64(sBu[1], Bu, Hc, n0, (kt + 1) * 32);
    a_tile_rows64(sBn[1], Bn, Hc, n0, (kt + 1) * 32);
    mm_step64x2(sA[0], sBu[0], sBn[0], wm, wn, cu, cn);
    wait_async0();
    __syncthreads();
    a_tile_gather128(sA[0], XL, Hc, rows, cnt, mbase, (kt + 2) * 32);
    a_tile_rows64(sBu[0], Bu, Hc, n0, (kt + 2) * 32);
    a_tile_rows64(sBn[0], Bn, Hc, n0, (kt + 2) * 32);
    mm_step64x2(sA[1], sBu[1], sBn[1], wm, wn, cu, cn);
    wait_async0();
    __syncthreads();
  }
  a_tile_gather128(sA[1], XL, Hc, rows, cnt, mbase, (kt + 1) * 32);
  a_tile_rows64(sBu[1], Bu, Hc, n0, (kt + 1) * 32);
  a_tile_rows64(sBn[1], Bn, Hc, n0, (kt + 1) * 32);
  mm_step64x2(sA[0], sBu[0], sBn[0], wm, wn, cu, cn);
  wait_async0();
  __syncthreads();
  mm_step64x2(sA[1], sBu[1], sBn[1], wm, wn, cu, cn);
  int hl = lane >> 4, ll = lane & 15;
#pragma unroll
  for (int i = 0; i < 2; ++i)
#pragma unroll
    for (int j = 0; j < 2; ++j)
#pragma unroll
      for (int r = 0; r < 8; ++r) {
        int ml = wm * 32 + i * 16 + r + 8 * hl;
        int gm = mbase + ml;
        if (gm < cnt) {
          int n = n0 + wn * 32 + j * 16 + ll;
          float u = cu[i][j][r] + b_up[e * Ic + n];
          float g = 0.5f * u * (1.f + erff(u * 0.70710678118f));
          float nv = cn[i][j][r] + b_new[e * Ic + n];
          hbuf[(long)(off + gm) * Ic + n] = f2bf(g * nv);
        }
      }
}

// ---------------- expert down GEMM + weighted scatter; grid (H/64, 32, E) ------
__global__ void __launch_bounds__(256) k_expert_down(
    const unsigned short* __restrict__ hbuf, const unsigned short* __restrict__ WdnT,
    const float* __restrict__ b_down, const int* __restrict__ counts,
    const int* __restrict__ segoff, const int* __restrict__ segtok,
    const float* __restrict__ segw, float* __restrict__ moe) {
  int e = blockIdx.z;
  int cnt = counts[e];
  int mbase = blockIdx.y * 128;
  if (mbase >= cnt) return;
  __shared__ __align__(16) unsigned short sA[2][128 * LDP];
  __shared__ __align__(16) unsigned short sB[2][64 * LDP];
  int off = segoff[e];
  int n0 = blockIdx.x * 64;
  const unsigned short* Bt = WdnT + (long)e * Hc * Ic;
  const unsigned short* Ae = hbuf + (long)off * Ic;
  int wave = threadIdx.x >> 5, lane = threadIdx.x & 31;
  int wm = wave & 3, wn = wave >> 2;
  f32x8 c[2][2];
#pragma unroll
  for (int i = 0; i < 2; ++i)
#pragma unroll
    for (int j = 0; j < 2; ++j) c[i][j] = zero8();
  a_tile_clamp128(sA[0], Ae, Ic, mbase, cnt, 0);
  a_tile_rows64(sB[0], Bt, Ic, n0, 0);
  wait_async0();
  __syncthreads();
  const int nk = Ic / 32;   // 96, even
  int kt = 0;
  for (; kt + 2 < nk; kt += 2) {
    a_tile_clamp128(sA[1], Ae, Ic, mbase, cnt, (kt + 1) * 32);
    a_tile_rows64(sB[1], Bt, Ic, n0, (kt + 1) * 32);
    mm_step64(sA[0], sB[0], wm, wn, c);
    wait_async0();
    __syncthreads();
    a_tile_clamp128(sA[0], Ae, Ic, mbase, cnt, (kt + 2) * 32);
    a_tile_rows64(sB[0], Bt, Ic, n0, (kt + 2) * 32);
    mm_step64(sA[1], sB[1], wm, wn, c);
    wait_async0();
    __syncthreads();
  }
  a_tile_clamp128(sA[1], Ae, Ic, mbase, cnt, (kt + 1) * 32);
  a_tile_rows64(sB[1], Bt, Ic, n0, (kt + 1) * 32);
  mm_step64(sA[0], sB[0], wm, wn, c);
  wait_async0();
  __syncthreads();
  mm_step64(sA[1], sB[1], wm, wn, c);
  int hl = lane >> 4, ll = lane & 15;
#pragma unroll
  for (int i = 0; i < 2; ++i)
#pragma unroll
    for (int j = 0; j < 2; ++j)
#pragma unroll
      for (int r = 0; r < 8; ++r) {
        int ml = wm * 32 + i * 16 + r + 8 * hl;
        int gm = mbase + ml;
        if (gm < cnt) {
          int tk = segtok[off + gm];
          float w = segw[off + gm];
          int n = n0 + wn * 32 + j * 16 + ll;
          float val = w * (c[i][j][r] + b_down[e * Hc + n]);
          atomicAdd(&moe[(long)tk * Hc + n], val);
        }
      }
}

__global__ void __launch_bounds__(256) k_add(const float* __restrict__ moe,
                                             const float* __restrict__ ao,
                                             float* __restrict__ out) {
  long i = (long)blockIdx.x * 256 + threadIdx.x;
  if (i < (long)NTOK * Hc) out[i] = moe[i] + ao[i];
}

// ---------------- host launch ----------------
extern "C" void kernel_launch(void* const* d_in, const int* in_sizes, int n_in,
                              void* d_out, int out_size, void* d_ws, size_t ws_size,
                              hipStream_t stream) {
  const float* x = (const float*)d_in[0];
  const float* Wq = (const float*)d_in[1];
  const float* bq = (const float*)d_in[2];
  const float* Wk = (const float*)d_in[3];
  const float* bk = (const float*)d_in[4];
  const float* Wv = (const float*)d_in[5];
  const float* bv = (const float*)d_in[6];
  const float* Wo = (const float*)d_in[7];
  const float* bo = (const float*)d_in[8];
  const float* ln_attn_g = (const float*)d_in[9];
  const float* ln_attn_b = (const float*)d_in[10];
  const float* ln_ffn_g = (const float*)d_in[11];
  const float* ln_ffn_b = (const float*)d_in[12];
  const float* Wr = (const float*)d_in[13];
  const float* br = (const float*)d_in[14];
  const float* W_up = (const float*)d_in[15];
  const float* b_up = (const float*)d_in[16];
  const float* W_new = (const float*)d_in[17];
  const float* b_new = (const float*)d_in[18];
  const float* W_down = (const float*)d_in[19];
  const float* b_down = (const float*)d_in[20];

  float* out_layer = (float*)d_out;
  float* out_rlog = (float*)d_out + (long)NTOK * Hc;

  char* w = (char*)d_ws;
  auto alloc = [&](size_t bytes) -> void* {
    void* p = (void*)w;
    w += (bytes + 255) & ~(size_t)255;
    return p;
  };
  unsigned short* X16 = (unsigned short*)alloc((size_t)NTOK * Hc * 2);
  unsigned short* QB  = (unsigned short*)alloc((size_t)NTOK * Hc * 2);
  unsigned short* KB  = (unsigned short*)alloc((size_t)NTOK * Hc * 2);
  unsigned short* VB  = (unsigned short*)alloc((size_t)NTOK * Hc * 2);
  unsigned short* CTX = (unsigned short*)alloc((size_t)NTOK * Hc * 2);
  unsigned short* XL  = (unsigned short*)alloc((size_t)NTOK * Hc * 2);
  unsigned short* WQT = (unsigned short*)alloc((size_t)Hc * Hc * 2);
  unsigned short* WKT = (unsigned short*)alloc((size_t)Hc * Hc * 2);
  unsigned short* WVT = (unsigned short*)alloc((size_t)Hc * Hc * 2);
  unsigned short* WOT = (unsigned short*)alloc((size_t)Hc * Hc * 2);
  unsigned short* WUPT = (unsigned short*)alloc((size_t)Ec * Ic * Hc * 2);
  unsigned short* WNWT = (unsigned short*)alloc((size_t)Ec * Ic * Hc * 2);
  unsigned short* WDNT = (unsigned short*)alloc((size_t)Ec * Hc * Ic * 2);
  float* TMP = (float*)alloc((size_t)NTOK * Hc * 4);
  float* AO  = (float*)alloc((size_t)NTOK * Hc * 4);
  float* MOE = (float*)alloc((size_t)NTOK * Hc * 4);
  int* TOP2I = (int*)alloc((size_t)NTOK * 2 * 4);
  float* TOP2W = (float*)alloc((size_t)NTOK * 2 * 4);
  int* COUNTS = (int*)alloc(64);
  int* SEGOFF = (int*)alloc(64);
  int* CURSOR = (int*)alloc(64);
  int* SEGTOK = (int*)alloc((size_t)NTOK * 2 * 4);
  float* SEGW = (float*)alloc((size_t)NTOK * 2 * 4);
  unsigned short* HBUF = (unsigned short*)alloc((size_t)NTOK * 2 * Ic * 2);

  const long NE = (long)NTOK * Hc;            // 3145728
  const int GE = (int)((NE + 255) / 256);     // 12288

  k_zero<<<GE, 256, 0, stream>>>(MOE, COUNTS);
  k_cvt<<<GE, 256, 0, stream>>>(x, X16, NE);

  // weight transpose+convert
  k_tcvt<<<dim3(Hc / 32, Hc / 32, 1), 256, 0, stream>>>(Wq, WQT, Hc, Hc, 0, 0);
  k_tcvt<<<dim3(Hc / 32, Hc / 32, 1), 256, 0, stream>>>(Wk, WKT, Hc, Hc, 0, 0);
  k_tcvt<<<dim3(Hc / 32, Hc / 32, 1), 256, 0, stream>>>(Wv, WVT, Hc, Hc, 0, 0);
  k_tcvt<<<dim3(Hc / 32, Hc / 32, 1), 256, 0, stream>>>(Wo, WOT, Hc, Hc, 0, 0);
  k_tcvt<<<dim3(Ic / 32, Hc / 32, Ec), 256, 0, stream>>>(W_up, WUPT, Hc, Ic,
                                                         (long)Hc * Ic, (long)Hc * Ic);
  k_tcvt<<<dim3(Ic / 32, Hc / 32, Ec), 256, 0, stream>>>(W_new, WNWT, Hc, Ic,
                                                         (long)Hc * Ic, (long)Hc * Ic);
  k_tcvt<<<dim3(Hc / 32, Ic / 32, Ec), 256, 0, stream>>>(W_down, WDNT, Ic, Hc,
                                                         (long)Hc * Ic, (long)Hc * Ic);

  // QKV projections
  dim3 gqkv(Hc / 128, NTOK / 128);
  k_gemm_qkv<<<gqkv, 256, 0, stream>>>(X16, WQT, bq, QB);
  k_gemm_qkv<<<gqkv, 256, 0, stream>>>(X16, WKT, bk, KB);
  k_gemm_qkv<<<gqkv, 256, 0, stream>>>(X16, WVT, bv, VB);

  // attention
  k_attn<<<dim3(Sc / 128, NHc, Bc), 256, 0, stream>>>(QB, KB, VB, CTX);

  // output projection
  k_gemm_wo<<<gqkv, 256, 0, stream>>>(CTX, WOT, TMP);

  // LN(attn residual) + LN(ffn)
  k_ln2<<<NTOK, 256, 0, stream>>>(TMP, x, bo, ln_attn_g, ln_attn_b, ln_ffn_g,
                                  ln_ffn_b, AO, XL);

  // router + dispatch
  k_router<<<NTOK / 8, 256, 0, stream>>>(XL, Wr, br, out_rlog, TOP2I, TOP2W, COUNTS);
  k_scan<<<1, 32, 0, stream>>>(COUNTS, SEGOFF, CURSOR);
  k_scatter<<<NTOK / 256, 256, 0, stream>>>(TOP2I, TOP2W, SEGOFF, CURSOR, SEGTOK, SEGW);

  // experts (top-2 only via gathered segments)
  k_expert_upnew<<<dim3(Ic / 64, NTOK / 128, Ec), 256, 0, stream>>>(
      XL, WUPT, WNWT, b_up, b_new, COUNTS, SEGOFF, SEGTOK, HBUF);
  k_expert_down<<<dim3(Hc / 64, NTOK / 128, Ec), 256, 0, stream>>>(
      HBUF, WDNT, b_down, COUNTS, SEGOFF, SEGTOK, SEGW, MOE);

  // final residual add
  k_add<<<GE, 256, 0, stream>>>(MOE, AO, out_layer);
}
